// DecoderLayer_39857296507466
// MI455X (gfx1250) — matrix operand order repaired and checked by reading, not compiled, not run on hardware
//
#include <hip/hip_runtime.h>
#include <hip/hip_bf16.h>

typedef __attribute__((ext_vector_type(16))) _Float16 v16h;
typedef __attribute__((ext_vector_type(8)))  _Float16 v8h;
typedef __attribute__((ext_vector_type(8)))  float    v8f;

#define B_SZ   4
#define N_TOK  6400
#define DM     256
#define H_NUM  8
#define HD     32
#define DFF    1024
#define NF     8
#define M_ROWS (B_SZ * N_TOK)   /* 25600 */

// ---------------------------------------------------------------------------
// elementwise f32 -> f16 (4 per thread)
// ---------------------------------------------------------------------------
__global__ __launch_bounds__(256) void f32_to_f16_k(const float* __restrict__ in,
                                                    _Float16* __restrict__ out,
                                                    long n4) {
    long i = (long)blockIdx.x * 256 + threadIdx.x;
    if (i < n4) {
        float4 v = ((const float4*)in)[i];
        _Float16* o = out + i * 4;
        o[0] = (_Float16)v.x; o[1] = (_Float16)v.y;
        o[2] = (_Float16)v.z; o[3] = (_Float16)v.w;
    }
}

// ---------------------------------------------------------------------------
// weight prep: f32 [K][N] row-major -> f16 [N][K]   (small matrices)
// ---------------------------------------------------------------------------
__global__ __launch_bounds__(256) void transpose_to_f16_k(const float* __restrict__ in,
                                                          _Float16* __restrict__ out,
                                                          int Kd, int Nd) {
    int idx = blockIdx.x * 256 + threadIdx.x;
    if (idx < Kd * Nd) {
        int k = idx / Nd, n = idx % Nd;
        out[(size_t)n * Kd + k] = (_Float16)in[idx];
    }
}

// ---------------------------------------------------------------------------
// WMMA GEMM:  out[M,N] = epilogue( A16[M,K] * Bt16[N,K]^T )
//   epilogue: y = (acc + bias[col]) * scale; if act: ELU; y += resid; store f32/f16
//   block = 256 threads = 8 waves (4 x 2), wave tile 32x32, block tile 128x64
// ---------------------------------------------------------------------------
__global__ __launch_bounds__(256, 2)
void gemm_f16_wmma(const _Float16* __restrict__ A, const _Float16* __restrict__ Bt,
                   const float* __restrict__ bias, const float* __restrict__ resid,
                   float* __restrict__ outF, _Float16* __restrict__ outH,
                   int K, int N, float scale, int act) {
    const int lane = threadIdx.x & 31;
    const int wave = threadIdx.x >> 5;
    const int wm   = wave & 3;
    const int wn   = wave >> 2;
    const int m0   = blockIdx.x * 128 + wm * 32;
    const int n0   = blockIdx.y * 64  + wn * 32;
    const int lh   = lane >> 4;   // which half of the wave
    const int lm   = lane & 15;

    v8f acc00 = {}, acc01 = {}, acc10 = {}, acc11 = {};

    // A operand in this code's contiguous k order (B matches): lanes 0-15 row M=lm hold K 0-15,
    // lanes 16-31 hold K 16-31.
    const _Float16* a0p = A  + (size_t)(m0 + lm)      * K + lh * 16;
    const _Float16* a1p = A  + (size_t)(m0 + 16 + lm) * K + lh * 16;
    // B operand in the same contiguous k order as A: lanes 0-15 col N=lm hold K 0-15,
    // lanes 16-31 hold K 16-31.  Bt is [N][K] so K is contiguous per lane.
    const _Float16* b0p = Bt + (size_t)(n0 + lm)      * K + lh * 16;
    const _Float16* b1p = Bt + (size_t)(n0 + 16 + lm) * K + lh * 16;

    for (int kb = 0; kb < K; kb += 32) {
        v8h a0lo = *(const v8h*)(a0p + kb);
        v8h a0hi = *(const v8h*)(a0p + kb + 8);
        v8h a1lo = *(const v8h*)(a1p + kb);
        v8h a1hi = *(const v8h*)(a1p + kb + 8);
        v8h b0lo = *(const v8h*)(b0p + kb);
        v8h b0hi = *(const v8h*)(b0p + kb + 8);
        v8h b1lo = *(const v8h*)(b1p + kb);
        v8h b1hi = *(const v8h*)(b1p + kb + 8);
        v16h a0, a1, b0, b1;
        #pragma unroll
        for (int e = 0; e < 8; ++e) {
            a0[e] = a0lo[e]; a0[e + 8] = a0hi[e];
            a1[e] = a1lo[e]; a1[e + 8] = a1hi[e];
            b0[e] = b0lo[e]; b0[e + 8] = b0hi[e];
            b1[e] = b1lo[e]; b1[e + 8] = b1hi[e];
        }
        acc00 = __builtin_amdgcn_wmma_f32_16x16x32_f16(false, a0, false, b0, (short)0, acc00, false, false);
        acc01 = __builtin_amdgcn_wmma_f32_16x16x32_f16(false, a0, false, b1, (short)0, acc01, false, false);
        acc10 = __builtin_amdgcn_wmma_f32_16x16x32_f16(false, a1, false, b0, (short)0, acc10, false, false);
        acc11 = __builtin_amdgcn_wmma_f32_16x16x32_f16(false, a1, false, b1, (short)0, acc11, false, false);
    }

    v8f accs[2][2] = {{acc00, acc01}, {acc10, acc11}};
    #pragma unroll
    for (int i = 0; i < 2; ++i) {
        #pragma unroll
        for (int j = 0; j < 2; ++j) {
            int col = n0 + j * 16 + lm;
            float bcol = bias ? bias[col] : 0.0f;
            #pragma unroll
            for (int r = 0; r < 8; ++r) {
                // D layout: VGPR r -> row (r + 8*laneHalf), col = lane%16
                int row = m0 + i * 16 + lh * 8 + r;
                float y = (accs[i][j][r] + bcol) * scale;
                if (act) y = (y > 0.0f) ? y : (__expf(y) - 1.0f);   // ELU
                size_t idx = (size_t)row * N + col;
                if (resid) y += resid[idx];
                if (outF) outF[idx] = y;
                if (outH) outH[idx] = (_Float16)y;
            }
        }
    }
}

// ---------------------------------------------------------------------------
// FAVOR+ features: out[b,h,n,m] = exp(q.proj[:,m] - 0.5*|q|^2) / sqrt(NF)
// one thread per (b,h,n)
// ---------------------------------------------------------------------------
__global__ __launch_bounds__(256) void phi_k(const float* __restrict__ qbuf,
                                             const float* __restrict__ feat,
                                             float* __restrict__ out) {
    int t  = blockIdx.x * 256 + threadIdx.x;     // (b,h,n) flat, b*H+h major
    int n  = t % N_TOK;
    int bh = t / N_TOK;
    int h  = bh & (H_NUM - 1);
    int b  = bh / H_NUM;
    const float* q = qbuf + ((size_t)(b * N_TOK + n)) * DM + h * HD;
    float qv[HD];
    float nrm = 0.0f;
    #pragma unroll
    for (int k = 0; k < HD; ++k) { qv[k] = q[k]; nrm += qv[k] * qv[k]; }
    nrm *= 0.5f;
    float* o = out + (size_t)t * NF;
    #pragma unroll
    for (int m = 0; m < NF; ++m) {
        float d = 0.0f;
        #pragma unroll
        for (int k = 0; k < HD; ++k) d += qv[k] * feat[k * NF + m];
        o[m] = __expf(d - nrm) * 0.35355339059327373f;   // 1/sqrt(8)
    }
}

// ---------------------------------------------------------------------------
// kv[b,h,m,k] = sum_n pk[b,h,n,m] * v[b,n,h*32+k];  pksum[b,h,m] = sum_n pk
// grid = B*H*chunks, block 256 (thread = (m,k)), atomic accumulate
// ---------------------------------------------------------------------------
#define KV_CHUNKS 25
__global__ __launch_bounds__(256) void kv_k(const float* __restrict__ pk,
                                            const float* __restrict__ v,
                                            float* __restrict__ kv,
                                            float* __restrict__ pksum) {
    int blk   = blockIdx.x;
    int chunk = blk % KV_CHUNKS;
    int bh    = blk / KV_CHUNKS;
    int h = bh & (H_NUM - 1), b = bh / H_NUM;
    int t = threadIdx.x;
    int m = t >> 5, kk = t & 31;
    const int span = N_TOK / KV_CHUNKS;   // 256
    int n0 = chunk * span;
    float acc = 0.0f, accs = 0.0f;
    for (int n = n0; n < n0 + span; ++n) {
        float p  = pk[((size_t)bh * N_TOK + n) * NF + m];
        float vv = v[((size_t)(b * N_TOK + n)) * DM + h * HD + kk];
        acc  += p * vv;
        accs += p;
    }
    atomicAdd(&kv[(size_t)bh * (NF * HD) + t], acc);
    if (kk == 0) atomicAdd(&pksum[bh * NF + m], accs);
}

// ---------------------------------------------------------------------------
// attn16[row, h*32+k] = (sum_m pq*kv[m,k]) / (sum_m pq*pksum[m] + eps), f16 out
// ---------------------------------------------------------------------------
__global__ __launch_bounds__(256) void combine_k(const float* __restrict__ pq,
                                                 const float* __restrict__ kv,
                                                 const float* __restrict__ pksum,
                                                 _Float16* __restrict__ out16) {
    size_t id = (size_t)blockIdx.x * 256 + threadIdx.x;   // over M_ROWS*DM
    int c = (int)(id & (DM - 1));
    size_t row = id >> 8;
    int h = c >> 5, kk = c & 31;
    int n = (int)(row % N_TOK), b = (int)(row / N_TOK);
    int bh = b * H_NUM + h;
    const float* pqv = pq + ((size_t)bh * N_TOK + n) * NF;
    const float* kvp = kv + (size_t)bh * (NF * HD);
    const float* ps  = pksum + bh * NF;
    float den = 1e-6f, num = 0.0f;
    #pragma unroll
    for (int m = 0; m < NF; ++m) {
        float p = pqv[m];
        den += p * ps[m];
        num += p * kvp[m * HD + kk];
    }
    out16[id] = (_Float16)(num / den);
}

// ---------------------------------------------------------------------------
// layernorm over rows of [M,256]; one wave32 per row (8 rows/block);
// writes f32 and optional f16 copy for the next GEMM input
// ---------------------------------------------------------------------------
__global__ __launch_bounds__(256) void layernorm_k(const float* __restrict__ in,
                                                   const float* __restrict__ g,
                                                   const float* __restrict__ bt,
                                                   float* __restrict__ outF,
                                                   _Float16* __restrict__ outH) {
    int lane = threadIdx.x & 31;
    size_t row = (size_t)blockIdx.x * 8 + (threadIdx.x >> 5);
    const float* p = in + row * DM + lane * 8;
    float x[8];
    *(float4*)(x)     = *(const float4*)(p);
    *(float4*)(x + 4) = *(const float4*)(p + 4);
    float s = 0.0f;
    #pragma unroll
    for (int e = 0; e < 8; ++e) s += x[e];
    #pragma unroll
    for (int off = 16; off > 0; off >>= 1) s += __shfl_xor(s, off, 32);
    float mu = s * (1.0f / DM);
    float vs = 0.0f;
    #pragma unroll
    for (int e = 0; e < 8; ++e) { float d = x[e] - mu; vs += d * d; }
    #pragma unroll
    for (int off = 16; off > 0; off >>= 1) vs += __shfl_xor(vs, off, 32);
    float rs = rsqrtf(vs * (1.0f / DM) + 1e-6f);
    #pragma unroll
    for (int e = 0; e < 8; ++e) {
        int c = lane * 8 + e;
        float y = (x[e] - mu) * rs * g[c] + bt[c];
        outF[row * DM + c] = y;
        if (outH) outH[row * DM + c] = (_Float16)y;
    }
}

// ---------------------------------------------------------------------------
extern "C" void kernel_launch(void* const* d_in, const int* in_sizes, int n_in,
                              void* d_out, int out_size, void* d_ws, size_t ws_size,
                              hipStream_t stream) {
    (void)in_sizes; (void)n_in; (void)out_size; (void)ws_size;

    const float* x    = (const float*)d_in[0];
    const float* enc  = (const float*)d_in[1];
    const float* m1_qw = (const float*)d_in[2];
    const float* m1_qb = (const float*)d_in[3];
    const float* m1_kw = (const float*)d_in[4];
    const float* m1_kb = (const float*)d_in[5];
    const float* m1_vw = (const float*)d_in[6];
    const float* m1_vb = (const float*)d_in[7];
    const float* m1_ow = (const float*)d_in[8];
    const float* m1_ob = (const float*)d_in[9];
    const float* m1_ft = (const float*)d_in[10];
    const float* m2_qw = (const float*)d_in[11];
    const float* m2_qb = (const float*)d_in[12];
    const float* m2_kw = (const float*)d_in[13];
    const float* m2_kb = (const float*)d_in[14];
    const float* m2_vw = (const float*)d_in[15];
    const float* m2_vb = (const float*)d_in[16];
    const float* m2_ow = (const float*)d_in[17];
    const float* m2_ob = (const float*)d_in[18];
    const float* m2_ft = (const float*)d_in[19];
    const float* fw1   = (const float*)d_in[20];
    const float* fb1   = (const float*)d_in[21];
    const float* fw2   = (const float*)d_in[22];
    const float* fb2   = (const float*)d_in[23];
    const float* ln1g  = (const float*)d_in[24];
    const float* ln1b  = (const float*)d_in[25];
    const float* ln2g  = (const float*)d_in[26];
    const float* ln2b  = (const float*)d_in[27];
    const float* ln3g  = (const float*)d_in[28];
    const float* ln3b  = (const float*)d_in[29];

    char* ws = (char*)d_ws;
    size_t off = 0;
    auto alloc = [&](size_t bytes) -> void* {
        void* p = ws + off;
        off += (bytes + 255) & ~(size_t)255;
        return p;
    };

    const size_t M = M_ROWS;
    // f16 weight buffers (transposed to [N][K])
    _Float16* qw1T = (_Float16*)alloc(DM * DM * 2);
    _Float16* kw1T = (_Float16*)alloc(DM * DM * 2);
    _Float16* vw1T = (_Float16*)alloc(DM * DM * 2);
    _Float16* ow1T = (_Float16*)alloc(DM * DM * 2);
    _Float16* qw2T = (_Float16*)alloc(DM * DM * 2);
    _Float16* kw2T = (_Float16*)alloc(DM * DM * 2);
    _Float16* vw2T = (_Float16*)alloc(DM * DM * 2);
    _Float16* ow2T = (_Float16*)alloc(DM * DM * 2);
    _Float16* w1T  = (_Float16*)alloc((size_t)DM * DFF * 2);
    _Float16* w2T  = (_Float16*)alloc((size_t)DFF * DM * 2);
    // f16 activations
    _Float16* x16    = (_Float16*)alloc(M * DM * 2);
    _Float16* enc16  = (_Float16*)alloc(M * DM * 2);
    _Float16* o1_16  = (_Float16*)alloc(M * DM * 2);
    _Float16* o2_16  = (_Float16*)alloc(M * DM * 2);
    _Float16* attn16 = (_Float16*)alloc(M * DM * 2);
    _Float16* h16    = (_Float16*)alloc(M * DFF * 2);
    // f32 activations
    float* qf  = (float*)alloc(M * DM * 4);     // also reused as LN input 'tf'
    float* kf  = (float*)alloc(M * DM * 4);
    float* vf  = (float*)alloc(M * DM * 4);
    float* o1f = (float*)alloc(M * DM * 4);
    float* o2f = (float*)alloc(M * DM * 4);
    float* pq  = (float*)alloc((size_t)B_SZ * H_NUM * N_TOK * NF * 4);
    float* pk  = (float*)alloc((size_t)B_SZ * H_NUM * N_TOK * NF * 4);
    float* kv  = (float*)alloc((size_t)B_SZ * H_NUM * NF * HD * 4);   // 32768 B
    float* pks = (float*)alloc((size_t)B_SZ * H_NUM * NF * 4);        // contiguous after kv
    float* tf  = qf;   // alias: q is dead by the time the pre-LN buffer is written

    const float s_qk = 0.42044820762685725f;   // 32^-0.25
    const dim3 blk(256);
    const dim3 g_gemm256(M_ROWS / 128, DM / 64);
    const dim3 g_gemm1024(M_ROWS / 128, DFF / 64);
    const int g_cvt = (int)(M * DM / 1024);
    const int g_phi = B_SZ * H_NUM * N_TOK / 256;
    const int g_kv  = B_SZ * H_NUM * KV_CHUNKS;
    const int g_cmb = (int)(M * DM / 256);
    const int g_ln  = M_ROWS / 8;

    // ---- prep: f16 conversions + weight transposes ----
    f32_to_f16_k<<<g_cvt, blk, 0, stream>>>(x,   x16,   (long)(M * DM / 4));
    f32_to_f16_k<<<g_cvt, blk, 0, stream>>>(enc, enc16, (long)(M * DM / 4));
    transpose_to_f16_k<<<DM * DM / 256, blk, 0, stream>>>(m1_qw, qw1T, DM, DM);
    transpose_to_f16_k<<<DM * DM / 256, blk, 0, stream>>>(m1_kw, kw1T, DM, DM);
    transpose_to_f16_k<<<DM * DM / 256, blk, 0, stream>>>(m1_vw, vw1T, DM, DM);
    transpose_to_f16_k<<<DM * DM / 256, blk, 0, stream>>>(m1_ow, ow1T, DM, DM);
    transpose_to_f16_k<<<DM * DM / 256, blk, 0, stream>>>(m2_qw, qw2T, DM, DM);
    transpose_to_f16_k<<<DM * DM / 256, blk, 0, stream>>>(m2_kw, kw2T, DM, DM);
    transpose_to_f16_k<<<DM * DM / 256, blk, 0, stream>>>(m2_vw, vw2T, DM, DM);
    transpose_to_f16_k<<<DM * DM / 256, blk, 0, stream>>>(m2_ow, ow2T, DM, DM);
    transpose_to_f16_k<<<DM * DFF / 256, blk, 0, stream>>>(fw1, w1T, DM, DFF);
    transpose_to_f16_k<<<DM * DFF / 256, blk, 0, stream>>>(fw2, w2T, DFF, DM);

    // ---- stage A: self-attention (m1) ----
    gemm_f16_wmma<<<g_gemm256, blk, 0, stream>>>(x16, qw1T, m1_qb, nullptr, qf, nullptr, DM, DM, s_qk, 0);
    gemm_f16_wmma<<<g_gemm256, blk, 0, stream>>>(x16, kw1T, m1_kb, nullptr, kf, nullptr, DM, DM, s_qk, 0);
    gemm_f16_wmma<<<g_gemm256, blk, 0, stream>>>(x16, vw1T, m1_vb, nullptr, vf, nullptr, DM, DM, 1.0f, 0);
    phi_k<<<g_phi, blk, 0, stream>>>(qf, m1_ft, pq);
    phi_k<<<g_phi, blk, 0, stream>>>(kf, m1_ft, pk);
    hipMemsetAsync(kv, 0, (size_t)B_SZ * H_NUM * NF * (HD + 1) * 4, stream);
    kv_k<<<g_kv, blk, 0, stream>>>(pk, vf, kv, pks);
    combine_k<<<g_cmb, blk, 0, stream>>>(pq, kv, pks, attn16);
    gemm_f16_wmma<<<g_gemm256, blk, 0, stream>>>(attn16, ow1T, m1_ob, x, tf, nullptr, DM, DM, 1.0f, 0);
    layernorm_k<<<g_ln, blk, 0, stream>>>(tf, ln1g, ln1b, o1f, o1_16);

    // ---- stage B: cross-attention (m2) ----
    gemm_f16_wmma<<<g_gemm256, blk, 0, stream>>>(o1_16, qw2T, m2_qb, nullptr, qf, nullptr, DM, DM, s_qk, 0);
    gemm_f16_wmma<<<g_gemm256, blk, 0, stream>>>(enc16, kw2T, m2_kb, nullptr, kf, nullptr, DM, DM, s_qk, 0);
    gemm_f16_wmma<<<g_gemm256, blk, 0, stream>>>(enc16, vw2T, m2_vb, nullptr, vf, nullptr, DM, DM, 1.0f, 0);
    phi_k<<<g_phi, blk, 0, stream>>>(qf, m2_ft, pq);
    phi_k<<<g_phi, blk, 0, stream>>>(kf, m2_ft, pk);
    hipMemsetAsync(kv, 0, (size_t)B_SZ * H_NUM * NF * (HD + 1) * 4, stream);
    kv_k<<<g_kv, blk, 0, stream>>>(pk, vf, kv, pks);
    combine_k<<<g_cmb, blk, 0, stream>>>(pq, kv, pks, attn16);
    gemm_f16_wmma<<<g_gemm256, blk, 0, stream>>>(attn16, ow2T, m2_ob, o1f, tf, nullptr, DM, DM, 1.0f, 0);
    layernorm_k<<<g_ln, blk, 0, stream>>>(tf, ln2g, ln2b, o2f, o2_16);

    // ---- stage C: FFN ----
    gemm_f16_wmma<<<g_gemm1024, blk, 0, stream>>>(o2_16, w1T, fb1, nullptr, nullptr, h16, DM, DFF, 1.0f, 1);
    gemm_f16_wmma<<<g_gemm256, blk, 0, stream>>>(h16, w2T, fb2, o2f, tf, nullptr, DFF, DM, 1.0f, 0);
    layernorm_k<<<g_ln, blk, 0, stream>>>(tf, ln3g, ln3b, (float*)d_out, nullptr);
}
